// ConvBlock_88905823027689
// MI455X (gfx1250) — matrix operand order repaired and checked by reading, not compiled, not run on hardware
//
#include <hip/hip_runtime.h>

typedef __attribute__((ext_vector_type(16))) _Float16     v16h;
typedef __attribute__((ext_vector_type(8)))  float        v8f;
typedef __attribute__((ext_vector_type(4)))  unsigned int v4u;

#define EPSBN 1e-3f
#define DTC   0.2f

// ---------------- weight transpose + f32->f16: wT[n*576+k] = w[k*64+n] ----------------
__global__ void wprep_kernel(const float* __restrict__ wf, const float* __restrict__ wg,
                             _Float16* __restrict__ wtf, _Float16* __restrict__ wtg) {
    int i = blockIdx.x * blockDim.x + threadIdx.x;   // 0 .. 2*576*64
    if (i >= 2 * 576 * 64) return;
    const float* w = (i < 576 * 64) ? wf : wg;
    _Float16*   wt = (i < 576 * 64) ? wtf : wtg;
    int r = i % (576 * 64);
    int n = r / 576, k = r % 576;
    wt[n * 576 + k] = (_Float16)w[k * 64 + n];
}

// ---------------- pointwise bn+relu -> f16 (channels = last dim, C=64) ----------------
__global__ void bnrelu_f16_kernel(const float* __restrict__ x,
                                  const float* __restrict__ g, const float* __restrict__ b,
                                  const float* __restrict__ m, const float* __restrict__ v,
                                  _Float16* __restrict__ out, int n) {
    int i = blockIdx.x * blockDim.x + threadIdx.x;
    if (i >= n) return;
    int c = i & 63;
    float sc = g[c] * rsqrtf(v[c] + EPSBN);
    float t  = (x[i] - m[c]) * sc + b[c];
    out[i] = (_Float16)fmaxf(t, 0.f);
}

// ---------------- 3x3 same conv, 64->64, implicit GEMM via WMMA f16 ----------------
// Grid: 4096 blocks x 128 threads. Each wave: 16 pixels x 64 couts (4 accumulators).
// K = 576 = 9 taps * 64 cin; fully unrolled tap(9) x half(2) loop, 4 WMMAs per K-block.
__launch_bounds__(128)
__global__ void conv3x3_wmma_kernel(const _Float16* __restrict__ a,   // (16,128,128,64) f16
                                    const _Float16* __restrict__ wt,  // 64 x 576 f16 (transposed)
                                    float*    __restrict__ out32,     // raw f32 result
                                    _Float16* __restrict__ out16,     // optional fused bn+relu f16
                                    const float* __restrict__ bng, const float* __restrict__ bnb,
                                    const float* __restrict__ bnm, const float* __restrict__ bnv) {
    extern __shared__ char smem_raw[];
    _Float16* lw = (_Float16*)smem_raw;                 // 64*576 f16 = 73728 B

    // stage transposed weights into LDS (4608 x 16B)
    {
        const uint4* src = (const uint4*)wt;
        uint4*       dst = (uint4*)lw;
        #pragma unroll 4
        for (int i = threadIdx.x; i < 4608; i += 128) dst[i] = src[i];
    }
    __syncthreads();

    const int lane = threadIdx.x & 31;
    const int wave = threadIdx.x >> 5;
    const int n    = lane & 15;          // A row (pixel) / B,C column (cout)
    const int hcf  = lane >> 4;          // lane half: selects K sub-range
    const int koff = hcf * 8;            // A-frag K offset within 32-block

    const int pbase = blockIdx.x * 64 + wave * 16;
    const int p     = pbase + n;         // this lane's pixel (global, linearizes b,y,x)
    const int xx    = p & 127;
    const int yy    = (p >> 7) & 127;

    // own-pixel base address: always in-bounds fallback for padded taps
    const _Float16* pok = a + (size_t)p * 64 + koff;
    // this lane's B column base in LDS (cout = n within tile, K sub-range by lane half)
    const _Float16* lwp = lw + n * 576 + (hcf << 3);

    v8f acc[4] = {};

    #pragma unroll
    for (int tap = 0; tap < 9; ++tap) {
        const int dy = tap / 3 - 1, dxo = tap % 3 - 1;        // compile-time constants
        const int y2 = yy + dy, x2 = xx + dxo;
        const bool ok = (((unsigned)y2 < 128u) & ((unsigned)x2 < 128u));
        const unsigned msk = ok ? 0xFFFFFFFFu : 0u;
        const int delta = (dy * 128 + dxo) * 64;              // element offset of this tap
        const _Float16* ap = pok + (ok ? delta : 0);          // clamped, always valid

        #pragma unroll
        for (int half = 0; half < 2; ++half) {                // cin block 0..31 / 32..63
            // A fragment: two unconditional b128 loads, zeroed by mask for padded taps
            v4u L0 = *(const v4u*)(ap + half * 32);
            v4u L1 = *(const v4u*)(ap + half * 32 + 16);
            L0 &= msk;
            L1 &= msk;
            union { v4u u[2]; v16h h; } ua;
            ua.u[0] = L0; ua.u[1] = L1;
            const v16h af = ua.h;

            // load ALL 4 B fragments first (distinct registers), then 4 WMMAs back-to-back
            const int kb = tap * 64 + half * 32;
            v16h bf[4];
            #pragma unroll
            for (int nt = 0; nt < 4; ++nt) {
                const _Float16* wp = lwp + nt * (16 * 576) + kb;
                union { v4u u[2]; v16h h; } ub;
                ub.u[0] = *(const v4u*)(wp);
                ub.u[1] = *(const v4u*)(wp + 16);
                bf[nt] = ub.h;
            }
            #pragma unroll
            for (int nt = 0; nt < 4; ++nt)
                acc[nt] = __builtin_amdgcn_wmma_f32_16x16x32_f16(
                    false, af, false, bf[nt], (short)0, acc[nt], false, false);
        }
    }

    // Epilogue: C/D layout -> lane holds column ch for rows m = v + 8*half
    #pragma unroll
    for (int nt = 0; nt < 4; ++nt) {
        const int ch = nt * 16 + n;
        float sc = 0.f, sh = 0.f;
        if (out16) {
            sc = bng[ch] * rsqrtf(bnv[ch] + EPSBN);
            sh = bnb[ch] - bnm[ch] * sc;
        }
        #pragma unroll
        for (int v = 0; v < 8; ++v) {
            const int m  = v + (hcf << 3);
            const int pp = pbase + m;
            const float val = acc[nt][v];
            out32[(size_t)pp * 64 + ch] = val;
            if (out16)
                out16[(size_t)pp * 64 + ch] = (_Float16)fmaxf(val * sc + sh, 0.f);
        }
    }
}

// ---------------- fused Sobel + K=5 diffusion + bn_o/relu, one (b,y) row per block -------
// The diffusion stencil only references W-shifts and channel-shifts (both wrap inside one
// row of 128x64 floats), so all 5 iterations stay in LDS. 8 arrays x 32KB = 256KB dyn LDS.
__launch_bounds__(256)
__global__ void diffusion_kernel(const float* __restrict__ f,    // h0 = h = f
                                 const float* __restrict__ g,
                                 const float* __restrict__ og, const float* __restrict__ ob,
                                 const float* __restrict__ om, const float* __restrict__ ov,
                                 float* __restrict__ out) {
    extern __shared__ char smem_raw[];
    float* H0 = (float*)smem_raw;
    float* H1 = H0 + 8192;
    float* H2 = H1 + 8192;
    float* BX = H2 + 8192;
    float* BY = BX + 8192;
    float* EE = BY + 8192;
    float* AX = EE + 8192;
    float* FF = AX + 8192;

    const int row = blockIdx.x;                 // b*128 + y
    const int b = row >> 7, y = row & 127;
    const size_t rbase = (size_t)row * 8192;
    const int tid = threadIdx.x;
    const int c   = tid & 63;                   // fixed channel per thread (stride 256)

    // phase 1: stage f row (into FF, H0, H1) and raw g row (into AX)
    for (int j = tid; j < 8192; j += 256) {
        float fv = f[rbase + j];
        FF[j] = fv; H0[j] = fv; H1[j] = fv;
        AX[j] = g[rbase + j];
    }
    const int yp = (y == 127) ? 126 : y + 1;    // REFLECT in H
    const int ym = (y == 0)   ? 1   : y - 1;
    const size_t rp = (size_t)(b * 128 + yp) * 8192;
    const size_t rm = (size_t)(b * 128 + ym) * 8192;

    for (int j = tid; j < 8192; j += 256) H2[j] = f[rp + j];   // stage f[yp]
    __syncthreads();
    for (int j = tid; j < 8192; j += 256) {
        int w = j >> 6;
        int wm = (w == 0) ? 1 : w - 1, wq = (w == 127) ? 126 : w + 1;  // REFLECT in W
        float a0 = H2[wm * 64 + c], a1 = H2[(w << 6) + c], a2 = H2[wq * 64 + c];
        BX[j] = a0 + 2.f * a1 + a2;     // dy accumulator (+yp row)
        BY[j] = a2 - a0;                // dx accumulator (yp contribution)
    }
    __syncthreads();
    for (int j = tid; j < 8192; j += 256) H2[j] = f[rm + j];   // stage f[ym]
    __syncthreads();
    for (int j = tid; j < 8192; j += 256) {
        int w = j >> 6;
        int wm = (w == 0) ? 1 : w - 1, wq = (w == 127) ? 126 : w + 1;
        float a0 = H2[wm * 64 + c], a1 = H2[(w << 6) + c], a2 = H2[wq * 64 + c];
        float dyv = BX[j] - (a0 + 2.f * a1 + a2);
        float dxv = BY[j] + (a2 - a0) + 2.f * (FF[wq * 64 + c] - FF[wm * 64 + c]);
        BX[j] = DTC / (dyv * dyv * 0.25f + 1.f);    // Bx = Dx*DT (Dx from dy)
        BY[j] = DTC / (dxv * dxv * 0.25f + 1.f);    // By = Dy*DT (Dy from dx)
        // E = (ux + vy)*DT from raw g (wrap-around rolls)
        int wmw = ((w + 127) & 127) << 6, wpw = ((w + 1) & 127) << 6;
        int cm = (c + 63) & 63, cp = (c + 1) & 63;
        float ux  = 0.5f * (AX[wmw + c] - AX[wpw + c]);
        float vyv = 0.5f * (AX[(w << 6) + cm] - AX[(w << 6) + cp]);
        EE[j] = (ux + vyv) * DTC;
    }
    __syncthreads();
    for (int j = tid; j < 8192; j += 256) AX[j] *= DTC;        // Ax = Ay = g*DT

    // phase 3: K=5 diffusion iterations entirely in LDS
    float* h0 = H0; float* h = H1; float* hn = H2;
    for (int it = 0; it < 5; ++it) {
        __syncthreads();
        for (int j = tid; j < 8192; j += 256) {
            int w = j >> 6;
            float ax = AX[j], bx = BX[j], by = BY[j];
            float Dv = 1.f / (1.f + 2.f * bx + 2.f * by);
            int wmw = ((w + 127) & 127) << 6, wpw = ((w + 1) & 127) << 6;
            int cm = (c + 63) & 63, cp = (c + 1) & 63;
            float r = (1.f - 2.f * bx - 2.f * by) * h0[j]
                    - 2.f * EE[j] * h[j]
                    + (-ax + 2.f * bx) * h[wmw + c]
                    + ( ax + 2.f * bx) * h[wpw + c]
                    + (-ax + 2.f * by) * h[(w << 6) + cm]
                    + ( ax + 2.f * by) * h[(w << 6) + cp]
                    + 2.f * DTC * FF[j];
            hn[j] = Dv * r;
        }
        float* t = h0; h0 = h; h = hn; hn = t;
    }
    __syncthreads();

    // phase 4: bn_o + relu -> out
    float sc = og[c] * rsqrtf(ov[c] + EPSBN);
    float sh = ob[c] - om[c] * sc;
    for (int j = tid; j < 8192; j += 256)
        out[rbase + j] = fmaxf(h[j] * sc + sh, 0.f);
}

// ------------------------------------------------------------------------------------
extern "C" void kernel_launch(void* const* d_in, const int* in_sizes, int n_in,
                              void* d_out, int out_size, void* d_ws, size_t ws_size,
                              hipStream_t stream) {
    (void)in_sizes; (void)n_in; (void)out_size; (void)ws_size;
    const float* x     = (const float*)d_in[0];
    const float* f_w   = (const float*)d_in[1];
    const float* g_w   = (const float*)d_in[2];
    const float* bnf_g = (const float*)d_in[3];
    const float* bnf_b = (const float*)d_in[4];
    const float* bnf_m = (const float*)d_in[5];
    const float* bnf_v = (const float*)d_in[6];
    const float* bng_g = (const float*)d_in[7];
    const float* bng_b = (const float*)d_in[8];
    const float* bng_m = (const float*)d_in[9];
    const float* bng_v = (const float*)d_in[10];
    const float* bno_g = (const float*)d_in[11];
    const float* bno_b = (const float*)d_in[12];
    const float* bno_m = (const float*)d_in[13];
    const float* bno_v = (const float*)d_in[14];

    const size_t N = (size_t)16 * 128 * 128 * 64;    // 16,777,216 elems
    float* out_h = (float*)d_out;                    // first output: h
    float* out_g = out_h + N;                        // second output: g

    // workspace layout
    char* ws = (char*)d_ws;
    _Float16* a_h   = (_Float16*)(ws);                       // 32 MB  relu(bn_f(x)) f16
    _Float16* b_h   = (_Float16*)(ws + N * 2);               // 32 MB  relu(bn_g(f)) f16
    float*    f_buf = (float*)   (ws + N * 4);               // 64 MB  f (f32)
    _Float16* wtf   = (_Float16*)(ws + N * 8);               // 72 KB
    _Float16* wtg   = (_Float16*)(ws + N * 8 + 576 * 64 * 2);// 72 KB

    // 1. transpose+convert both weight tensors
    wprep_kernel<<<(2 * 576 * 64 + 255) / 256, 256, 0, stream>>>(f_w, g_w, wtf, wtg);

    // 2. a = relu(bn_f(x)) -> f16
    bnrelu_f16_kernel<<<(int)(N / 256), 256, 0, stream>>>(x, bnf_g, bnf_b, bnf_m, bnf_v,
                                                          a_h, (int)N);

    // 3. f = conv(a, f_w); fused epilogue also emits relu(bn_g(f)) as f16 for conv2
    conv3x3_wmma_kernel<<<4096, 128, 576 * 64 * 2, stream>>>(
        a_h, wtf, f_buf, b_h, bng_g, bng_b, bng_m, bng_v);

    // 4. g = conv(b, g_w) -> raw f32 straight into the second output slot
    conv3x3_wmma_kernel<<<4096, 128, 576 * 64 * 2, stream>>>(
        b_h, wtg, out_g, nullptr, nullptr, nullptr, nullptr, nullptr);

    // 5. sobel coeffs + 5 diffusion iterations + bn_o/relu, one (b,y) row per block
    diffusion_kernel<<<16 * 128, 256, 8 * 8192 * 4, stream>>>(
        f_buf, out_g, bno_g, bno_b, bno_m, bno_v, out_h);
}
